// SequenceHMM_23295902613712
// MI455X (gfx1250) — compile-verified, matrix operand order checked by reading, not run on hardware
//
#include <hip/hip_runtime.h>
#include <hip/hip_bf16.h>

// HMM forward: S=512 states, V=1024 obs symbols, B=64 sequences, T=512 steps.
// Recurrence converted to f16 WMMA GEMM via max-factored logsumexp:
//   alpha[b,:] = off[b] + a_rel[b,:],  P = exp(a_rel) in f16 (<= 1.0),
//   Et[i][j] = exp(trans[j][i]) in f16 (precomputed, L2-resident),
//   acc[b,i] = sum_j P[b,j]*E[j,i]  -> v_wmma_f32_16x16x32_f16,
//   r = log(acc) + (emission[i,obs]-lse[i]); off += rowmax(r); P = exp(r-max).

#define NSTATE 512
#define NOBS   1024
#define NBATCH 64
#define NSEQ   512
#define PSTRIDE 520   // 512 + 8 halfs pad; keeps 16B alignment (520*2 = 1040 = 65*16)

typedef __attribute__((ext_vector_type(16))) _Float16 v16h;
typedef __attribute__((ext_vector_type(8)))  _Float16 v8h;
typedef __attribute__((ext_vector_type(8)))  float    v8f;

// ---- kernel 1: per-state logsumexp of emission logits (log_softmax denominator)
__global__ void hmm_row_lse(const float* __restrict__ em, float* __restrict__ lse) {
  __shared__ float red[256];
  const int s = blockIdx.x;
  const float* row = em + (size_t)s * NOBS;
  float m = -INFINITY;
  for (int v = threadIdx.x; v < NOBS; v += 256) m = fmaxf(m, row[v]);
  red[threadIdx.x] = m;
  __syncthreads();
  for (int off = 128; off > 0; off >>= 1) {
    if (threadIdx.x < off) red[threadIdx.x] = fmaxf(red[threadIdx.x], red[threadIdx.x + off]);
    __syncthreads();
  }
  m = red[0];
  __syncthreads();
  float ssum = 0.f;
  for (int v = threadIdx.x; v < NOBS; v += 256) ssum += __expf(row[v] - m);
  red[threadIdx.x] = ssum;
  __syncthreads();
  for (int off = 128; off > 0; off >>= 1) {
    if (threadIdx.x < off) red[threadIdx.x] += red[threadIdx.x + off];
    __syncthreads();
  }
  if (threadIdx.x == 0) lse[s] = m + __logf(red[0]);
}

// ---- kernel 2: Et[i][j] = (f16) exp(trans[j][i])   (transposed so B-frags load
//      like A-frags of B^T per the 16-bit WMMA lane layout)
__global__ void hmm_build_et(const float* __restrict__ trans, _Float16* __restrict__ Et) {
  const int idx = blockIdx.x * blockDim.x + threadIdx.x;   // 0 .. S*S-1
  const int i = idx >> 9;
  const int j = idx & (NSTATE - 1);
  Et[idx] = (_Float16)__expf(trans[j * NSTATE + i]);
}

// ---- kernel 3: the serial forward recurrence. 4 workgroups x 16 batch rows.
//      1024 threads = 32 wave32; wave n computes output columns [16n, 16n+16).
__global__ __launch_bounds__(1024)
void hmm_forward(const float* __restrict__ prior,
                 const float* __restrict__ emission,
                 const int*   __restrict__ obs,
                 const _Float16* __restrict__ Et,
                 const float* __restrict__ lse,
                 float* __restrict__ out) {
  __shared__ __align__(16) _Float16 P[16][PSTRIDE];  // p = exp(alpha_rel), f16
  __shared__ float wred[32][16];                     // per-wave partial max/sum
  __shared__ float mred[16];                         // per-b row max this step
  __shared__ float offb[16];                         // running log-offset per b

  const int tid  = threadIdx.x;
  const int lane = tid & 31;
  const int wave = tid >> 5;        // N-tile index 0..31
  const int hl   = lane & 15;       // half-lane
  const int quad = lane >> 4;       // 0/1: which K-chunk (A/B) and M-half (C)
  const int bg0  = blockIdx.x * 16; // first global batch row of this workgroup

  const int i = wave * 16 + hl;     // output state column owned by this lane (C N-index)
  const float lse_i   = lse[i];
  const float prior_i = prior[i];
  const float* em_i   = emission + (size_t)i * NOBS;
  const _Float16* bt  = Et + (size_t)i * NSTATE;   // row i of Et == column i of E
  const int b_base = quad * 8;      // C rows: b = b_base + rr (VGPR rr of the C frag)

  float r[8];

  for (int t = 0; t < NSEQ; ++t) {
    if (t > 0) {
      // acc[16x512] = P @ E ; this wave's 16x16 tile, K = 512 in 16 WMMA steps.
      v8f acc = {};
      const int abase = quad * 8;   // K sub-offset per 16-bit A-layout
#pragma unroll
      for (int k = 0; k < 16; ++k) {
        const int kb = k * 32 + abase;
        union { v16h v; v8h h[2]; } a, bf;
        // A-frag: lane holds row M=hl, K = kb..kb+7 and kb+16..kb+23 (LDS, 2x b128)
        a.h[0] = *(const v8h*)&P[hl][kb];
        a.h[1] = *(const v8h*)&P[hl][kb + 16];
        // B-frag: A-layout of B^T -> row i of Et, same K chunks (global, 2x b128)
        bf.h[0] = *(const v8h*)(bt + kb);
        bf.h[1] = *(const v8h*)(bt + kb + 16);
        if (k < 15) __builtin_prefetch(bt + kb + 32, 0, 1);  // global_prefetch_b8
        acc = __builtin_amdgcn_wmma_f32_16x16x32_f16(
            /*neg_a=*/false, a.v, /*neg_b=*/false, bf.v,
            /*c_mod=*/(short)0, acc, /*reuse_a=*/false, /*reuse_b=*/false);
      }
      // r[b,i] = log(acc) + emission[i, obs[b,t]] - lse[i]
#pragma unroll
      for (int rr = 0; rr < 8; ++rr) {
        const int b = b_base + rr;
        const int o = obs[(size_t)(bg0 + b) * NSEQ + t];
        r[rr] = __logf(acc[rr]) + em_i[o] - lse_i;
      }
    } else {
      // alpha0 = prior + log_softmax(emission)[i, obs[b,0]]
#pragma unroll
      for (int rr = 0; rr < 8; ++rr) {
        const int b = b_base + rr;
        const int o = obs[(size_t)(bg0 + b) * NSEQ];
        r[rr] = prior_i + em_i[o] - lse_i;
      }
    }

    // per-wave row max over this tile's 16 columns (reduce across 16-lane group)
    float mx[8];
#pragma unroll
    for (int rr = 0; rr < 8; ++rr) {
      float v = r[rr];
      for (int off = 8; off > 0; off >>= 1) v = fmaxf(v, __shfl_xor(v, off, 16));
      mx[rr] = v;
    }
    if (hl == 0) {
#pragma unroll
      for (int rr = 0; rr < 8; ++rr) wred[wave][b_base + rr] = mx[rr];
    }
    __syncthreads();
    if (tid < 16) {                       // cross-wave max + offset update
      float m = wred[0][tid];
      for (int w = 1; w < 32; ++w) m = fmaxf(m, wred[w][tid]);
      mred[tid] = m;
      offb[tid] = (t == 0) ? m : (offb[tid] + m);
    }
    __syncthreads();
    // P = exp(r - rowmax), f16, max element is exactly 1.0
#pragma unroll
    for (int rr = 0; rr < 8; ++rr) {
      const int b = b_base + rr;
      P[b][i] = (_Float16)__expf(r[rr] - mred[b]);
    }
    __syncthreads();                      // P complete before next step's WMMA reads
  }

  // out[b] = off[b] + log( sum_i P[b,i] )
  float ps[8];
#pragma unroll
  for (int rr = 0; rr < 8; ++rr) {
    float v = (float)P[b_base + rr][i];
    for (int off = 8; off > 0; off >>= 1) v += __shfl_xor(v, off, 16);
    ps[rr] = v;
  }
  if (hl == 0) {
#pragma unroll
    for (int rr = 0; rr < 8; ++rr) wred[wave][b_base + rr] = ps[rr];
  }
  __syncthreads();
  if (tid < 16) {
    float ssum = 0.f;
    for (int w = 0; w < 32; ++w) ssum += wred[w][tid];
    out[bg0 + tid] = offb[tid] + __logf(ssum);
  }
}

extern "C" void kernel_launch(void* const* d_in, const int* in_sizes, int n_in,
                              void* d_out, int out_size, void* d_ws, size_t ws_size,
                              hipStream_t stream) {
  const float* prior    = (const float*)d_in[0];           // [512]
  const float* trans    = (const float*)d_in[1];           // [512,512]
  const float* emission = (const float*)d_in[2];           // [512,1024]
  const int*   obsv     = (const int*)d_in[3];             // [64,512]
  float* out = (float*)d_out;                              // [64] f32

  _Float16* Et = (_Float16*)d_ws;                                        // 512 KB
  float* lse   = (float*)((char*)d_ws + (size_t)NSTATE * NSTATE * sizeof(_Float16));

  hmm_row_lse<<<NSTATE, 256, 0, stream>>>(emission, lse);
  hmm_build_et<<<(NSTATE * NSTATE) / 256, 256, 0, stream>>>(trans, Et);
  hmm_forward<<<NBATCH / 16, 1024, 0, stream>>>(prior, emission, obsv, Et, lse, out);
}
